// TrainableKMeans_48155173323353
// MI455X (gfx1250) — compile-verified, matrix operand order checked by reading, not run on hardware
//
#include <hip/hip_runtime.h>
#include <hip/hip_bf16.h>
#include <stdint.h>

// Problem constants (from reference)
#define NTOK   32768      // B*S
#define INDIM  512
#define DDIM   512
#define NCLUS  4096

typedef __attribute__((ext_vector_type(16))) __bf16    v16bf;
typedef __attribute__((ext_vector_type(8)))  __bf16    v8bf;
typedef __attribute__((ext_vector_type(2)))  __bf16    v2bf;
typedef __attribute__((ext_vector_type(8)))  float     v8f;
typedef __attribute__((ext_vector_type(4)))  unsigned  v4u;
typedef __attribute__((ext_vector_type(8)))  int       v8i;
typedef __attribute__((ext_vector_type(4)))  int       v4i;

__device__ __forceinline__ v8f wmma_bf16(v16bf a, v16bf b, v8f c) {
  // D = A(16x32 bf16) * B(32x16 bf16) + C(16x16 f32)
  return __builtin_amdgcn_wmma_f32_16x16x32_bf16(
      /*neg_a=*/false, a, /*neg_b=*/false, b,
      /*c_mod=*/(short)0, c, /*reuse_a=*/false, /*reuse_b=*/false);
}

// ---------------------------------------------------------------------------
// Kernel 0: zero the h_sq accumulator (ws is poisoned 0xAA by harness)
// ---------------------------------------------------------------------------
__global__ __launch_bounds__(256) void zero_hsq_kernel(float* __restrict__ hsq) {
  int i = blockIdx.x * 256 + threadIdx.x;
  if (i < NTOK) hsq[i] = 0.0f;
}

// ---------------------------------------------------------------------------
// Kernel 1: per-cluster prep.  One wave per cluster.
//   cw[c]  = centroids[c,:] . W2[:,0]
//   csq[c] = ||centroids[c,:]||^2
//   cb[c,:] = bf16(centroids[c,:])   (row-major == WMMA B-operand layout)
// ---------------------------------------------------------------------------
__global__ __launch_bounds__(256) void prep_kernel(
    const float* __restrict__ cent, const float* __restrict__ W2,
    __bf16* __restrict__ cb, float* __restrict__ cw, float* __restrict__ csq) {
  int lane = threadIdx.x & 31;
  int c = blockIdx.x * 8 + (threadIdx.x >> 5);     // 512 blocks * 8 waves = 4096
  const float* row = cent + (size_t)c * DDIM + lane * 16;
  float dot = 0.f, sq = 0.f;
  v16bf t;
#pragma unroll
  for (int i = 0; i < 16; ++i) {
    float v = row[i];
    dot += v * W2[lane * 16 + i];
    sq  += v * v;
    t[i] = (__bf16)v;
  }
  *(v16bf*)(cb + (size_t)c * DDIM + lane * 16) = t;
#pragma unroll
  for (int m = 16; m >= 1; m >>= 1) {
    dot += __shfl_xor(dot, m, 32);
    sq  += __shfl_xor(sq,  m, 32);
  }
  if (lane == 0) { cw[c] = dot; csq[c] = sq; }
}

// ---------------------------------------------------------------------------
// Kernel 2: h = x @ W1 + b1   (bf16 WMMA, f32 accumulate)
//   grid = (NTOK/64, DDIM/128); block = 256 (8 waves: 4 M-tiles x 2 N-halves)
//   writes hb (bf16 h, row-major) and atomically accumulates hsq.
// ---------------------------------------------------------------------------
__global__ __launch_bounds__(256) void embed_kernel(
    const float* __restrict__ x, const float* __restrict__ W1,
    const float* __restrict__ b1, __bf16* __restrict__ hb,
    float* __restrict__ hsq) {
  __shared__ __align__(64) __bf16 sA[64 * 32];     // [row][k]  (A layout)
  __shared__ __align__(64) __bf16 sB[128 * 32];    // [n][k]    (B layout, transposed)
  int tid  = threadIdx.x;
  int lane = tid & 31, wave = tid >> 5;
  int wm = wave & 3, wn = wave >> 2;
  int tokbase = blockIdx.x * 64;
  int nbase   = blockIdx.y * 128;

  v8f acc[4];
#pragma unroll
  for (int nt = 0; nt < 4; ++nt) acc[nt] = (v8f){0,0,0,0,0,0,0,0};

  for (int k0 = 0; k0 < INDIM; k0 += 32) {
    {   // stage A: 64 rows x 32 k; 8 contiguous k per thread -> one ds_store_b128
      int r = tid >> 2, kc = (tid & 3) * 8;
      const float* src = x + (size_t)(tokbase + r) * INDIM + k0 + kc;
      v8bf t8;
#pragma unroll
      for (int i = 0; i < 8; ++i) t8[i] = (__bf16)src[i];
      *(v8bf*)&sA[r * 32 + kc] = t8;
    }
    {   // stage B transposed: thread owns 2 consecutive k x 8 n -> 8 ds_store_b32
      int kk = (tid & 15) * 2, n0 = (tid >> 4) * 8;
      const float* s0 = W1 + (size_t)(k0 + kk) * DDIM + nbase + n0;
      const float* s1 = s0 + DDIM;
#pragma unroll
      for (int j = 0; j < 8; ++j) {
        v2bf p;
        p[0] = (__bf16)s0[j];
        p[1] = (__bf16)s1[j];
        *(v2bf*)&sB[(n0 + j) * 32 + kk] = p;
      }
    }
    __syncthreads();

    int kh = (lane >> 4) * 16;
    v16bf a = *(const v16bf*)&sA[(wm * 16 + (lane & 15)) * 32 + kh];
#pragma unroll
    for (int nt = 0; nt < 4; ++nt) {
      v16bf b = *(const v16bf*)&sB[(wn * 64 + nt * 16 + (lane & 15)) * 32 + kh];
      acc[nt] = wmma_bf16(a, b, acc[nt]);
    }
    __syncthreads();
  }

  // epilogue: +bias, store bf16 h, accumulate row sums of squares
  int colbase = nbase + wn * 64;
  int rowbase = tokbase + wm * 16 + (lane >> 4) * 8;
  float sq[8];
#pragma unroll
  for (int v = 0; v < 8; ++v) sq[v] = 0.f;
#pragma unroll
  for (int nt = 0; nt < 4; ++nt) {
    int col = colbase + nt * 16 + (lane & 15);
    float bias = b1[col];
#pragma unroll
    for (int v = 0; v < 8; ++v) {
      float val = acc[nt][v] + bias;
      hb[(size_t)(rowbase + v) * DDIM + col] = (__bf16)val;
      sq[v] += val * val;
    }
  }
#pragma unroll
  for (int v = 0; v < 8; ++v)
#pragma unroll
    for (int m = 8; m >= 1; m >>= 1) sq[v] += __shfl_xor(sq[v], m, 32);
  if ((lane & 15) == 0) {
#pragma unroll
    for (int v = 0; v < 8; ++v) atomicAdd(&hsq[rowbase + v], sq[v]);
  }
}

// ---------------------------------------------------------------------------
// Kernel 3: fused dist -> exp2 -> weighted reduction.
//   grid = NTOK/128, block = 256.  Wave w owns 16 token rows; streams all
//   4096 clusters in chunks of 32 (two 16x16 WMMA N-tiles sharing the A
//   fragment; two independent accumulator chains).  h tile (128x512 bf16 =
//   128KB) loaded into LDS by the Tensor Data Mover.
// ---------------------------------------------------------------------------
__global__ __launch_bounds__(256) void cluster_kernel(
    const __bf16* __restrict__ hb, const __bf16* __restrict__ cb,
    const float* __restrict__ cw, const float* __restrict__ csq,
    const float* __restrict__ hsq, const float* __restrict__ b2,
    float* __restrict__ out) {
  extern __shared__ __bf16 sH[];                   // [128][512] bf16 = 128 KB
  int tid = threadIdx.x, lane = tid & 31, wave = tid >> 5;
  int tokbase = blockIdx.x * 128;

#if defined(__has_builtin) && __has_builtin(__builtin_amdgcn_tensor_load_to_lds)
  if (wave == 0) {
    // Tensor DMA descriptor (CDNA5 ISA):
    // 2D tensor hb[NTOK][512] bf16; tile = 128 rows x 512 cols at tokbase.
    unsigned lds_off = (unsigned)(uintptr_t)(&sH[0]);     // LDS aperture: [31:0] = offset
    uint64_t ga = (uint64_t)(uintptr_t)(hb + (size_t)tokbase * DDIM);
    v4u g0;
    g0[0] = 1u;                                        // count=1, is_restore=0, gather=0
    g0[1] = lds_off;                                   // lds_addr
    g0[2] = (unsigned)ga;                              // global_addr[31:0]
    g0[3] = (unsigned)(ga >> 32) | (2u << 30);         // global_addr[56:32] | type=2
    v8i g1;
    g1[0] = (1 << 16);                 // workgroup_mask=0, data_size=1 (2B)
    g1[1] = (int)(512u << 16);         // tensor_dim0[15:0]=512 (elements per row)
    g1[2] = (int)0x80000000u;          // tensor_dim1[15:0]=32768
    g1[3] = (int)(512u << 16);         // tile_dim0=512
    g1[4] = 128;                       // tile_dim1=128, tile_dim2=0
    g1[5] = 512;                       // tensor_dim0_stride[31:0]=512
    g1[6] = 0;                         // stride hi, dim1_stride lo
    g1[7] = 0;                         // dim1_stride hi (unused for 2D)
    v4i gz4 = (v4i){0, 0, 0, 0};       // groups 2/3 unused (2D tensor)
    v8i gz8 = (v8i){0, 0, 0, 0, 0, 0, 0, 0};
    __builtin_amdgcn_tensor_load_to_lds(g0, g1, gz4, gz4, gz8, 0);
    __builtin_amdgcn_s_wait_tensorcnt(0);
  }
#else
  {   // fallback: cooperative 16B copies
    const uint4* src = (const uint4*)(hb + (size_t)tokbase * DDIM);
    uint4* dst = (uint4*)sH;
    for (int i = tid; i < (128 * DDIM * 2) / 16; i += 256) dst[i] = src[i];
  }
#endif
  __syncthreads();

  int kh = (lane >> 4) * 16;
  // Keep all 16 K-step A fragments of this wave's 16 rows resident in VGPRs.
  v16bf afrag[16];
#pragma unroll
  for (int ks = 0; ks < 16; ++ks)
    afrag[ks] = *(const v16bf*)&sH[(wave * 16 + (lane & 15)) * DDIM + ks * 32 + kh];

  int rowbase = tokbase + wave * 16 + (lane >> 4) * 8;
  float hs[8], num[8], den[8];
#pragma unroll
  for (int v = 0; v < 8; ++v) {
    hs[v] = hsq[rowbase + v];
    num[v] = 0.f;
    den[v] = 0.f;
  }

  for (int c0 = 0; c0 < NCLUS; c0 += 32) {
    int cl0 = c0 + (lane & 15);
    int cl1 = cl0 + 16;
    const __bf16* bp0 = cb + (size_t)cl0 * DDIM + kh;
    const __bf16* bp1 = cb + (size_t)cl1 * DDIM + kh;
    v8f acc0 = (v8f){0,0,0,0,0,0,0,0};
    v8f acc1 = (v8f){0,0,0,0,0,0,0,0};
#pragma unroll
    for (int ks = 0; ks < 16; ++ks) {
      v16bf b0 = *(const v16bf*)(bp0 + ks * 32);
      v16bf b1v = *(const v16bf*)(bp1 + ks * 32);
      acc0 = wmma_bf16(afrag[ks], b0, acc0);     // two independent chains,
      acc1 = wmma_bf16(afrag[ks], b1v, acc1);    // same A fragment
    }
    float cs0 = csq[cl0], wc0 = cw[cl0];
    float cs1 = csq[cl1], wc1 = cw[cl1];
#pragma unroll
    for (int v = 0; v < 8; ++v) {
      float d20 = fmaxf(hs[v] + cs0 - 2.0f * acc0[v], 0.0f);
      float d21 = fmaxf(hs[v] + cs1 - 2.0f * acc1[v], 0.0f);
      float p0 = exp2f(__builtin_amdgcn_sqrtf(d20) * -1.4426950408889634f);
      float p1 = exp2f(__builtin_amdgcn_sqrtf(d21) * -1.4426950408889634f);
      den[v] += p0 + p1;
      num[v] += p0 * wc0 + p1 * wc1;
    }
  }

  // reduce partial num/den across each 16-lane half (columns of the chunk)
#pragma unroll
  for (int v = 0; v < 8; ++v) {
#pragma unroll
    for (int m = 8; m >= 1; m >>= 1) {
      num[v] += __shfl_xor(num[v], m, 32);
      den[v] += __shfl_xor(den[v], m, 32);
    }
  }
  if ((lane & 15) == 0) {
    float bb = b2[0];
    const float inv_sqrtD = 0.04419417382415922f;     // 1/sqrt(512)
#pragma unroll
    for (int v = 0; v < 8; ++v)
      out[rowbase + v] = num[v] / den[v] * inv_sqrtD + bb;
  }
}

// ---------------------------------------------------------------------------
extern "C" void kernel_launch(void* const* d_in, const int* in_sizes, int n_in,
                              void* d_out, int out_size, void* d_ws, size_t ws_size,
                              hipStream_t stream) {
  const float* x    = (const float*)d_in[0];
  const float* W1   = (const float*)d_in[1];
  const float* b1   = (const float*)d_in[2];
  const float* cent = (const float*)d_in[3];
  const float* W2   = (const float*)d_in[4];
  const float* b2   = (const float*)d_in[5];
  float* out = (float*)d_out;

  // workspace layout (all offsets 256B aligned)
  char* ws = (char*)d_ws;
  __bf16* cb  = (__bf16*)(ws);                                   // 4 MB
  float*  cw  = (float*)(ws + (size_t)4 * 1024 * 1024);          // 16 KB
  float*  csq = (float*)(ws + (size_t)4 * 1024 * 1024 + 16384);  // 16 KB
  float*  hsq = (float*)(ws + (size_t)4 * 1024 * 1024 + 32768);  // 128 KB
  __bf16* hb  = (__bf16*)(ws + (size_t)4 * 1024 * 1024 + 32768 + 131072);  // 32 MB

  zero_hsq_kernel<<<dim3(NTOK / 256), dim3(256), 0, stream>>>(hsq);
  prep_kernel<<<dim3(NCLUS / 8), dim3(256), 0, stream>>>(cent, W2, cb, cw, csq);
  embed_kernel<<<dim3(NTOK / 64, DDIM / 128), dim3(256), 0, stream>>>(x, W1, b1, hb, hsq);
  cluster_kernel<<<dim3(NTOK / 128), dim3(256), 128 * DDIM * 2, stream>>>(
      hb, cb, cw, csq, hsq, b2, out);
}
